// MultiHeadSelfAttention_32658931319464
// MI455X (gfx1250) — matrix operand order repaired and checked by reading, not compiled, not run on hardware
//
#include <hip/hip_runtime.h>
#include <hip/hip_bf16.h>

#define DIM     256
#define NH      8
#define HD      32
#define BATCH   4
#define SEQ     4096            // 64*64
#define MROWS   (BATCH*SEQ)     // 16384

typedef __bf16 bf16;
typedef __attribute__((ext_vector_type(16))) __bf16 v16bf;
typedef __attribute__((ext_vector_type(8)))  __bf16 v8bf;
typedef __attribute__((ext_vector_type(8)))  float  v8f;
typedef __attribute__((ext_vector_type(4)))  unsigned int v4u;
typedef __attribute__((ext_vector_type(8)))  int    v8i;
typedef __attribute__((ext_vector_type(4)))  int    v4i;

#ifndef __has_builtin
#define __has_builtin(x) 0
#endif
#define HAVE_TDM  __has_builtin(__builtin_amdgcn_tensor_load_to_lds)
#define HAVE_TCNT __has_builtin(__builtin_amdgcn_s_wait_tensorcnt)

static __device__ __forceinline__ v8f wmma_bf16(v16bf a, v16bf b, v8f c) {
    return __builtin_amdgcn_wmma_f32_16x16x32_bf16(false, a, false, b, (short)0, c,
                                                   false, false);
}

// A-fragment (16x32 bf16): lane holds row (lane&15); K = {kb..kb+15},
// kb = lane<16 ? 0 : 16. p = row base + kb. Two 16B loads.
static __device__ __forceinline__ v16bf load_afrag(const bf16* p) {
    v8bf lo = *(const v8bf*)(p);
    v8bf hi = *(const v8bf*)(p + 8);
    return __builtin_shufflevector(lo, hi, 0,1,2,3,4,5,6,7,8,9,10,11,12,13,14,15);
}
// B-fragment: lane holds 16 contiguous elements; two 16B loads (16B-align safe).
static __device__ __forceinline__ v16bf load_bfrag(const bf16* p) {
    v8bf lo = *(const v8bf*)(p);
    v8bf hi = *(const v8bf*)(p + 8);
    return __builtin_shufflevector(lo, hi, 0,1,2,3,4,5,6,7,8,9,10,11,12,13,14,15);
}

// ---------------------------------------------------------------------------
// Kernel 1: convert weights fp32 -> bf16
// ---------------------------------------------------------------------------
__global__ __launch_bounds__(256) void wconv_kernel(const float* __restrict__ wq,
                                                    const float* __restrict__ wp,
                                                    bf16* __restrict__ wqb,
                                                    bf16* __restrict__ wpb) {
    int i = blockIdx.x * 256 + threadIdx.x;
    if (i < 3 * DIM * DIM) wqb[i] = (bf16)wq[i];
    if (i < DIM * DIM)     wpb[i] = (bf16)wp[i];
}

// ---------------------------------------------------------------------------
// Kernel 2: x (B,C,S) fp32 -> xb (B*S, C) bf16 via LDS tile transpose
// ---------------------------------------------------------------------------
__global__ __launch_bounds__(256) void xpose_kernel(const float* __restrict__ x,
                                                    bf16* __restrict__ xb) {
    __shared__ float t[32][33];
    int b  = blockIdx.z;
    int c0 = blockIdx.y * 32;
    int s0 = blockIdx.x * 32;
    int tx = threadIdx.x, ty = threadIdx.y;          // block (32,8)
#pragma unroll
    for (int i = 0; i < 32; i += 8)
        t[ty + i][tx] = x[((size_t)b * DIM + c0 + ty + i) * SEQ + s0 + tx];
    __syncthreads();
    bf16* op = xb + ((size_t)b * SEQ + s0) * DIM + c0;
#pragma unroll
    for (int i = 0; i < 32; i += 8)
        op[(size_t)(ty + i) * DIM + tx] = (bf16)t[tx][ty + i];
}

// ---------------------------------------------------------------------------
// Kernel 3: QKV GEMM (16384x256)x(768x256)^T + bias -> Q,K,(V^T) bf16
// ---------------------------------------------------------------------------
__global__ __launch_bounds__(256) void qkv_kernel(const bf16* __restrict__ X,
                                                  const bf16* __restrict__ W,
                                                  const float* __restrict__ bias,
                                                  bf16* __restrict__ Q,
                                                  bf16* __restrict__ Kd,
                                                  bf16* __restrict__ Vt) {
    const int NT = (3 * DIM) / 16;                   // 48
    int lane = threadIdx.x & 31;
    int wave = threadIdx.x >> 5;
    int tile = blockIdx.x * 8 + wave;
    int mt = tile / NT, nt = tile % NT;

    int arow = lane & 15, akb = (lane < 16) ? 0 : 16;
    int ncol = lane & 15, nkb = (lane < 16) ? 0 : 16;

    const bf16* ap = X + (size_t)(mt * 16 + arow) * DIM + akb;
    const bf16* bp = W + (size_t)(nt * 16 + ncol) * DIM + nkb;

    v8f acc = {};
#pragma unroll
    for (int k0 = 0; k0 < DIM; k0 += 32) {
        acc = wmma_bf16(load_afrag(ap + k0), load_bfrag(bp + k0), acc);
    }

    int j = nt * 16 + ncol;
    float bj = bias[j];
    int mm = j >> 8, cc = j & 255;
    int head = cc >> 5, dd = cc & 31;
#pragma unroll
    for (int r = 0; r < 8; ++r) {
        int row = mt * 16 + ((lane < 16) ? r : r + 8);
        int b_ = row >> 12, s_ = row & (SEQ - 1);
        bf16 val = (bf16)(acc[r] + bj);
        size_t bh = (size_t)(b_ * NH + head);
        if (mm == 0)      Q [(bh * SEQ + s_) * HD + dd] = val;
        else if (mm == 1) Kd[(bh * SEQ + s_) * HD + dd] = val;
        else              Vt[(bh * HD + dd) * SEQ + s_] = val;
    }
}

// ---------------------------------------------------------------------------
// TDM staging of one 32-key tile of K (32x32, contiguous) and V^T (32x32,
// stride SEQ) into LDS, with pad_enable giving an 80B row stride (40 elems).
// ---------------------------------------------------------------------------
#if HAVE_TDM
static __device__ __forceinline__ void tdm_load(unsigned int lds_off, const bf16* gsrc,
                                                unsigned int tensor_d0, unsigned int tensor_d1,
                                                unsigned int tile_d0, unsigned int tile_d1,
                                                unsigned long long stride0) {
    unsigned long long ga = (unsigned long long)(uintptr_t)gsrc;
    v4u g0;
    g0.x = 1u;                                       // count=1 (valid descriptor)
    g0.y = lds_off;                                  // LDS byte address
    g0.z = (unsigned int)ga;                         // global addr -> bits 95:64
    g0.w = (unsigned int)(ga >> 32) | 0x80000000u;   // addr[56:32] | type=2
    v8i g1;
    // data_size=2B | pad_enable | pad_interval=16 DW | pad_amount=4 DW
    g1[0] = (int)((1u << 16) | (1u << 20) | (3u << 22) | (3u << 25));
    g1[1] = (int)((tensor_d0 & 0xFFFFu) << 16);
    g1[2] = (int)((tensor_d0 >> 16) | ((tensor_d1 & 0xFFFFu) << 16));
    g1[3] = (int)((tensor_d1 >> 16) | (tile_d0 << 16));
    g1[4] = (int)tile_d1;                            // tile_dim2 = 0
    g1[5] = (int)(unsigned int)(stride0 & 0xFFFFFFFFu);
    g1[6] = (int)(unsigned int)((stride0 >> 32) & 0xFFFFu);
    g1[7] = 0;
    v4i z4 = {0, 0, 0, 0};
    v8i z8 = {0, 0, 0, 0, 0, 0, 0, 0};
    // 6-arg form (clang-23 / therock headers): groups 2,3 unused, extra group, cpol
    __builtin_amdgcn_tensor_load_to_lds(g0, g1, z4, z4, z8, 0);
}
#endif

static __device__ __forceinline__ void stage_tiles(bf16* kdst, bf16* vdst,
                                                   const bf16* Kb, const bf16* Vb,
                                                   int kt, int lane) {
#if HAVE_TDM
    (void)lane;
    // K tile: 2048 contiguous bytes, 1-D copy + LDS padding every 64B
    tdm_load((unsigned int)(uintptr_t)kdst, Kb + (size_t)kt * HD,
             1024u, 1u, 1024u, 1u, 1024ull);
    // V tile: 32 rows (d) x 32 keys, global row stride SEQ elements
    tdm_load((unsigned int)(uintptr_t)vdst, Vb + kt,
             (unsigned int)SEQ, 32u, 32u, 32u, (unsigned long long)SEQ);
#else
    // cooperative fallback: wave0 lanes copy 16B chunks, padded 40-elem rows
#pragma unroll
    for (int c = lane; c < 128; c += 32) {
        int row = c >> 2, off = (c & 3) * 8;
        *(v8bf*)(kdst + row * 40 + off) =
            *(const v8bf*)(Kb + (size_t)(kt + row) * HD + off);
        *(v8bf*)(vdst + row * 40 + off) =
            *(const v8bf*)(Vb + (size_t)row * SEQ + kt + off);
    }
#endif
}

// ---------------------------------------------------------------------------
// Kernel 4: flash attention. 8 waves/block; each wave owns 32 query rows.
// K/V tiles staged to LDS by TDM (double-buffered, TENSORcnt + barriers).
// Per 32-key tile: 4 score WMMAs + online softmax + LDS relayout + 4 PV WMMAs.
// ---------------------------------------------------------------------------
__global__ __launch_bounds__(256) void attn_kernel(const bf16* __restrict__ Q,
                                                   const bf16* __restrict__ K,
                                                   const bf16* __restrict__ Vt,
                                                   bf16* __restrict__ O) {
    __shared__ __align__(128) bf16 kbuf[2][32 * 40];
    __shared__ __align__(128) bf16 vbuf[2][32 * 40];
    __shared__ __align__(128) bf16 pbuf[8][32 * 40];
    int lane = threadIdx.x & 31;
    int wave = threadIdx.x >> 5;
    int b = blockIdx.z, head = blockIdx.y;
    int bh = b * NH + head;
    int qbase = blockIdx.x * 256 + wave * 32;

    const bf16* Qb = Q  + (size_t)bh * SEQ * HD;
    const bf16* Kb = K  + (size_t)bh * SEQ * HD;
    const bf16* Vb = Vt + (size_t)bh * HD * SEQ;

    int arow = lane & 15, akb = (lane < 16) ? 0 : 16;
    int ncol = lane & 15, nkb = (lane < 16) ? 0 : 16;

    v16bf qa0 = load_afrag(Qb + (size_t)(qbase + arow) * HD + akb);
    v16bf qa1 = load_afrag(Qb + (size_t)(qbase + 16 + arow) * HD + akb);

    float m[2][8], l[2][8];
    v8f o00 = {}, o01 = {}, o10 = {}, o11 = {};
#pragma unroll
    for (int q = 0; q < 2; ++q)
#pragma unroll
        for (int r = 0; r < 8; ++r) { m[q][r] = -1e30f; l[q][r] = 0.0f; }
    const float scale = 0.17677669529663687f;        // 1/sqrt(32)
    bf16* pw = &pbuf[wave][0];

    if (wave == 0) stage_tiles(&kbuf[0][0], &vbuf[0][0], Kb, Vb, 0, lane);

    for (int kt = 0, it = 0; kt < SEQ; kt += 32, ++it) {
        int buf = it & 1;
        if (wave == 0) {
            if (kt + 32 < SEQ) {
                stage_tiles(&kbuf[buf ^ 1][0], &vbuf[buf ^ 1][0], Kb, Vb, kt + 32, lane);
#if HAVE_TDM
#if HAVE_TCNT
                __builtin_amdgcn_s_wait_tensorcnt(2);
#else
                asm volatile("s_wait_tensorcnt 0x2" ::: "memory");
#endif
#endif
            }
#if HAVE_TDM
            else {
#if HAVE_TCNT
                __builtin_amdgcn_s_wait_tensorcnt(0);
#else
                asm volatile("s_wait_tensorcnt 0x0" ::: "memory");
#endif
            }
#endif
        }
        __syncthreads();                              // publish current buf

        const bf16* kls = &kbuf[buf][0];
        const bf16* vls = &vbuf[buf][0];
        v16bf kf0 = load_bfrag(kls + ncol * 40 + nkb);         // keys 0..15
        v16bf kf1 = load_bfrag(kls + (16 + ncol) * 40 + nkb);  // keys 16..31
        v16bf vf0 = load_bfrag(vls + ncol * 40 + nkb);         // d 0..15
        v16bf vf1 = load_bfrag(vls + (16 + ncol) * 40 + nkb);  // d 16..31

#pragma unroll
        for (int q = 0; q < 2; ++q) {
            v8f s0 = {}, s1 = {};
            s0 = wmma_bf16(q ? qa1 : qa0, kf0, s0);
            s1 = wmma_bf16(q ? qa1 : qa0, kf1, s1);
#pragma unroll
            for (int r = 0; r < 8; ++r) {
                float a0 = s0[r] * scale, a1 = s1[r] * scale;
                float mx = fmaxf(a0, a1);
#pragma unroll
                for (int d = 1; d < 16; d <<= 1) mx = fmaxf(mx, __shfl_xor(mx, d, 32));
                float mnew = fmaxf(m[q][r], mx);
                float p0 = __expf(a0 - mnew);
                float p1 = __expf(a1 - mnew);
                float rs = p0 + p1;
#pragma unroll
                for (int d = 1; d < 16; d <<= 1) rs += __shfl_xor(rs, d, 32);
                float cf = __expf(m[q][r] - mnew);
                m[q][r] = mnew;
                l[q][r] = l[q][r] * cf + rs;
                if (q == 0) { o00[r] *= cf; o01[r] *= cf; }
                else        { o10[r] *= cf; o11[r] *= cf; }
                int prow = q * 16 + ((lane < 16) ? r : r + 8);
                pw[prow * 40 + ncol]      = (bf16)p0;
                pw[prow * 40 + 16 + ncol] = (bf16)p1;
            }
        }
        // same-wave DS is in-order; stop compiler reordering + drain DS
        asm volatile("s_wait_dscnt 0x0" ::: "memory");
        v16bf pa0 = load_afrag(pw + arow * 40 + akb);
        v16bf pa1 = load_afrag(pw + (16 + arow) * 40 + akb);
        o00 = wmma_bf16(pa0, vf0, o00);
        o01 = wmma_bf16(pa0, vf1, o01);
        o10 = wmma_bf16(pa1, vf0, o10);
        o11 = wmma_bf16(pa1, vf1, o11);
        __syncthreads();                              // done reading buf
    }

    // normalize; write O in (B*S, C) bf16 layout, C = head*32 + d
#pragma unroll
    for (int q = 0; q < 2; ++q) {
#pragma unroll
        for (int r = 0; r < 8; ++r) {
            int srow = qbase + q * 16 + ((lane < 16) ? r : r + 8);
            float inv = 1.0f / l[q][r];
            size_t base = ((size_t)b * SEQ + srow) * DIM + head * HD;
            float va = (q == 0) ? o00[r] : o10[r];
            float vb = (q == 0) ? o01[r] : o11[r];
            O[base + ncol]      = (bf16)(va * inv);
            O[base + 16 + ncol] = (bf16)(vb * inv);
        }
    }
}

// ---------------------------------------------------------------------------
// Kernel 5: proj GEMM (16384x256)x(256x256)^T + bias -> out (B,C,S) fp32
// ---------------------------------------------------------------------------
__global__ __launch_bounds__(256) void proj_kernel(const bf16* __restrict__ A,
                                                   const bf16* __restrict__ W,
                                                   const float* __restrict__ bias,
                                                   float* __restrict__ out) {
    const int NT = DIM / 16;
    int lane = threadIdx.x & 31;
    int wave = threadIdx.x >> 5;
    int tile = blockIdx.x * 8 + wave;
    int mt = tile / NT, nt = tile % NT;

    int arow = lane & 15, akb = (lane < 16) ? 0 : 16;
    int ncol = lane & 15, nkb = (lane < 16) ? 0 : 16;

    const bf16* ap = A + (size_t)(mt * 16 + arow) * DIM + akb;
    const bf16* bp = W + (size_t)(nt * 16 + ncol) * DIM + nkb;

    v8f acc = {};
#pragma unroll
    for (int k0 = 0; k0 < DIM; k0 += 32) {
        acc = wmma_bf16(load_afrag(ap + k0), load_bfrag(bp + k0), acc);
    }

    int j = nt * 16 + ncol;
    float bj = bias[j];
#pragma unroll
    for (int r = 0; r < 8; ++r) {
        int row = mt * 16 + ((lane < 16) ? r : r + 8);
        int b_ = row >> 12, s_ = row & (SEQ - 1);
        out[((size_t)b_ * DIM + j) * SEQ + s_] = acc[r] + bj;
    }
}

// ---------------------------------------------------------------------------
extern "C" void kernel_launch(void* const* d_in, const int* in_sizes, int n_in,
                              void* d_out, int out_size, void* d_ws, size_t ws_size,
                              hipStream_t stream) {
    (void)in_sizes; (void)n_in; (void)out_size; (void)ws_size;
    const float* x      = (const float*)d_in[0];
    const float* w_qkv  = (const float*)d_in[1];
    const float* b_qkv  = (const float*)d_in[2];
    const float* w_proj = (const float*)d_in[3];
    const float* b_proj = (const float*)d_in[4];
    float* out = (float*)d_out;

    char* ws = (char*)d_ws;
    const size_t SZ_MAT = (size_t)MROWS * DIM * sizeof(bf16);       // 8 MB
    bf16* xb  = (bf16*)(ws);                  ws += SZ_MAT;
    bf16* Qd  = (bf16*)(ws);                  ws += SZ_MAT;
    bf16* Kd  = (bf16*)(ws);                  ws += SZ_MAT;
    bf16* Vt  = (bf16*)(ws);                  ws += SZ_MAT;
    bf16* Od  = (bf16*)(ws);                  ws += SZ_MAT;
    bf16* wqb = (bf16*)(ws);                  ws += (size_t)3 * DIM * DIM * sizeof(bf16);
    bf16* wpb = (bf16*)(ws);

    wconv_kernel<<<(3 * DIM * DIM) / 256, 256, 0, stream>>>(w_qkv, w_proj, wqb, wpb);
    xpose_kernel<<<dim3(SEQ / 32, DIM / 32, BATCH), dim3(32, 8), 0, stream>>>(x, xb);
    qkv_kernel<<<(MROWS / 16) * (3 * DIM / 16) / 8, 256, 0, stream>>>(xb, wqb, b_qkv,
                                                                      Qd, Kd, Vt);
    attn_kernel<<<dim3(SEQ / 256, NH, BATCH), 256, 0, stream>>>(Qd, Kd, Vt, Od);
    proj_kernel<<<(MROWS / 16) * (DIM / 16) / 8, 256, 0, stream>>>(Od, wpb, b_proj, out);
}
